// InfoNCELoss_7507602833781
// MI455X (gfx1250) — matrix operand order repaired and checked by reading, not compiled, not run on hardware
//
#include <hip/hip_runtime.h>
#include <hip/hip_bf16.h>

typedef _Float16 v16h __attribute__((ext_vector_type(16)));
typedef _Float16 v8h  __attribute__((ext_vector_type(8)));
typedef float    v8f  __attribute__((ext_vector_type(8)));
typedef int      v4i  __attribute__((vector_size(4 * sizeof(int))));

// async-copy pointer casts: builtin wants (v4i AS1*, v4i AS3*, imm, imm)
#define ASYNC_GPTR(x) ((__attribute__((address_space(1))) v4i*)(x))
#define ASYNC_LPTR(x) ((__attribute__((address_space(3))) v4i*)(x))

#define BS    8192
#define DIM   768
#define BM    128
#define BN    128
#define BK    32
#define LDK   40              // padded LDS row stride (halves): 20 dwords -> conflict-free 16-row gathers
#define KCH   (DIM / BK)      // 24 K-chunks
#define NBLK  (BS / BM)       // 64 blocks per dim

// ---------------------------------------------------------------------------
// Kernel 1: row L2-normalize fp32 -> f16  (one block per row, 256 thr, 3 elts/thr)
// ---------------------------------------------------------------------------
__global__ __launch_bounds__(256) void nrm_kernel(const float* __restrict__ src,
                                                  _Float16* __restrict__ dst) {
  const int row = blockIdx.x;
  const int tid = threadIdx.x;
  const float* r = src + (size_t)row * DIM;
  float x0 = r[tid], x1 = r[tid + 256], x2 = r[tid + 512];
  float s = x0 * x0 + x1 * x1 + x2 * x2;
#pragma unroll
  for (int off = 16; off > 0; off >>= 1) s += __shfl_down(s, off, 32);
  __shared__ float red[8];
  if ((tid & 31) == 0) red[tid >> 5] = s;
  __syncthreads();
  float tot = red[0] + red[1] + red[2] + red[3] + red[4] + red[5] + red[6] + red[7];
  float inv = 1.0f / fmaxf(sqrtf(tot), 1e-8f);
  _Float16* d = dst + (size_t)row * DIM;
  d[tid]       = (_Float16)(x0 * inv);
  d[tid + 256] = (_Float16)(x1 * inv);
  d[tid + 512] = (_Float16)(x2 * inv);
}

// ---------------------------------------------------------------------------
// Kernel 2: fused WMMA GEMM + streaming exp-sum reductions.
// Double-buffered LDS fed by GLOBAL_LOAD_ASYNC_TO_LDS_B128 (ASYNCcnt pipeline):
//   issue async copies for chunk k+1, s_wait_asynccnt<=4 (chunk k landed),
//   barrier, 8x back-to-back WMMA on chunk k, barrier.
// Block = 128x128 tile of S; 8 waves x (16x128 strip). S never materialized.
// ---------------------------------------------------------------------------
__global__ __launch_bounds__(256) void gemm_kernel(const _Float16* __restrict__ tN,
                                                   const _Float16* __restrict__ cN,
                                                   float* __restrict__ rowPartial,
                                                   float* __restrict__ colPartial,
                                                   float* __restrict__ diag) {
  __shared__ __align__(16) _Float16 aT[2][BM * LDK];   // 2 x 10 KB
  __shared__ __align__(16) _Float16 bT[2][BN * LDK];   // 2 x 10 KB
  __shared__ float rowPart[BM];
  __shared__ float colPart[BN];

  const int tid  = threadIdx.x;
  const int wave = tid >> 5;
  const int lane = tid & 31;
  const int r    = lane & 15;     // fragment row / N index
  const int half = lane >> 4;     // lane half selects K sub-block
  const int bx   = blockIdx.x;    // col block
  const int by   = blockIdx.y;    // row block
  const int rowBase = by * BM;
  const int colBase = bx * BN;

  v8f acc[8] = {};                // wave's 16x128 strip

  // each thread owns two 16B segments of each 128x32 tile
  const int idx0 = tid, idx1 = tid + 256;
  const int row0 = idx0 >> 2, cc0 = (idx0 & 3) * 8;
  const int row1 = idx1 >> 2, cc1 = (idx1 & 3) * 8;

  auto issue = [&](int kc, int buf) {
    const int k0 = kc * BK;
    const _Float16* ga0 = tN + (size_t)(rowBase + row0) * DIM + k0 + cc0;
    const _Float16* gb0 = cN + (size_t)(colBase + row0) * DIM + k0 + cc0;
    const _Float16* ga1 = tN + (size_t)(rowBase + row1) * DIM + k0 + cc1;
    const _Float16* gb1 = cN + (size_t)(colBase + row1) * DIM + k0 + cc1;
    __builtin_amdgcn_global_load_async_to_lds_b128(
        ASYNC_GPTR(ga0), ASYNC_LPTR(&aT[buf][row0 * LDK + cc0]), 0, 0);
    __builtin_amdgcn_global_load_async_to_lds_b128(
        ASYNC_GPTR(gb0), ASYNC_LPTR(&bT[buf][row0 * LDK + cc0]), 0, 0);
    __builtin_amdgcn_global_load_async_to_lds_b128(
        ASYNC_GPTR(ga1), ASYNC_LPTR(&aT[buf][row1 * LDK + cc1]), 0, 0);
    __builtin_amdgcn_global_load_async_to_lds_b128(
        ASYNC_GPTR(gb1), ASYNC_LPTR(&bT[buf][row1 * LDK + cc1]), 0, 0);
  };

  union frag { v16h v; v8h h[2]; };

  auto compute = [&](int buf) {
    const _Float16* Ab = &aT[buf][0];
    const _Float16* Bb = &bT[buf][0];
    // A fragment (16x32 f16): half0 K{0..7,16..23}, half1 K{8..15,24..31}
    frag af;
    af.h[0] = *(const v8h*)(Ab + (wave * 16 + r) * LDK + half * 8);
    af.h[1] = *(const v8h*)(Ab + (wave * 16 + r) * LDK + 16 + half * 8);
    // B fragments (32x16 f16): half0 K{0..7,16..23}, half1 K{8..15,24..31}
    frag bf[8];
#pragma unroll
    for (int nt = 0; nt < 8; ++nt) {
      bf[nt].h[0] = *(const v8h*)(Bb + (nt * 16 + r) * LDK + half * 8);
      bf[nt].h[1] = *(const v8h*)(Bb + (nt * 16 + r) * LDK + 16 + half * 8);
    }
#pragma unroll
    for (int nt = 0; nt < 8; ++nt)
      acc[nt] = __builtin_amdgcn_wmma_f32_16x16x32_f16(
          false, af.v, false, bf[nt].v, (short)0, acc[nt], false, false);
  };

  // ---- software pipeline ----
  issue(0, 0);
#pragma unroll 2
  for (int kc = 0; kc < KCH; ++kc) {
    if (kc + 1 < KCH) {
      issue(kc + 1, (kc + 1) & 1);
      asm volatile("s_wait_asynccnt 0x4" ::: "memory");   // chunk kc landed (in-order)
    } else {
      asm volatile("s_wait_asynccnt 0x0" ::: "memory");
    }
    __syncthreads();            // all waves' copies of chunk kc visible
    compute(kc & 1);
    __syncthreads();            // buffer reusable for kc+2 issue
  }

  // ---- epilogue: exp(S-1) row/col reductions + diagonal capture ----
  if (tid < BM) { rowPart[tid] = 0.0f; colPart[tid] = 0.0f; }
  __syncthreads();

  float rowAcc[8];
#pragma unroll
  for (int v = 0; v < 8; ++v) rowAcc[v] = 0.0f;

#pragma unroll
  for (int nt = 0; nt < 8; ++nt) {
    float colAcc = 0.0f;
#pragma unroll
    for (int v = 0; v < 8; ++v) {
      // C/D layout: element v on this lane is (M = v + 8*half, N = r)
      float sv = acc[nt][v];
      float e  = __expf(sv - 1.0f);       // cos/T <= 1 -> safe single-pass LSE
      rowAcc[v] += e;
      colAcc    += e;
      int gRow = rowBase + wave * 16 + v + 8 * half;
      int gCol = colBase + nt * 16 + r;
      if (gRow == gCol) diag[gRow] = sv;
    }
    atomicAdd(&colPart[nt * 16 + r], colAcc);
  }
#pragma unroll
  for (int v = 0; v < 8; ++v)
    atomicAdd(&rowPart[wave * 16 + v + 8 * half], rowAcc[v]);
  __syncthreads();

  if (tid < BM) {
    rowPartial[(size_t)bx * BS + rowBase + tid] = rowPart[tid];
    colPartial[(size_t)by * BS + colBase + tid] = colPart[tid];
  }
}

// ---------------------------------------------------------------------------
// Kernel 3: per-row loss  loss_i = log(rs)+1 + log(cs)+1 - 2*diag_i
// ---------------------------------------------------------------------------
__global__ __launch_bounds__(256) void loss_kernel(const float* __restrict__ rowPartial,
                                                   const float* __restrict__ colPartial,
                                                   const float* __restrict__ diag,
                                                   float* __restrict__ lossArr) {
  int i = blockIdx.x * 256 + threadIdx.x;
  float rs = 0.0f, cs = 0.0f;
  for (int b = 0; b < NBLK; ++b) {
    rs += rowPartial[(size_t)b * BS + i];
    cs += colPartial[(size_t)b * BS + i];
  }
  lossArr[i] = logf(rs) + logf(cs) + 2.0f - 2.0f * diag[i];
}

// ---------------------------------------------------------------------------
// Kernel 4: mean over 8192 -> scalar
// ---------------------------------------------------------------------------
__global__ __launch_bounds__(256) void reduce_kernel(const float* __restrict__ lossArr,
                                                     float* __restrict__ out) {
  int tid = threadIdx.x;
  float s = 0.0f;
  for (int i = tid; i < BS; i += 256) s += lossArr[i];
#pragma unroll
  for (int off = 16; off > 0; off >>= 1) s += __shfl_down(s, off, 32);
  __shared__ float red[8];
  if ((tid & 31) == 0) red[tid >> 5] = s;
  __syncthreads();
  if (tid == 0) {
    float t = 0.0f;
    for (int w = 0; w < 8; ++w) t += red[w];
    out[0] = t / (float)BS;
  }
}

// ---------------------------------------------------------------------------
extern "C" void kernel_launch(void* const* d_in, const int* in_sizes, int n_in,
                              void* d_out, int out_size, void* d_ws, size_t ws_size,
                              hipStream_t stream) {
  const float* text = (const float*)d_in[0];
  const float* ctr  = (const float*)d_in[1];
  float* out = (float*)d_out;

  char* ws = (char*)d_ws;
  size_t off = 0;
  auto carve = [&](size_t bytes) -> char* {
    char* p = ws + off;
    off += (bytes + 255) & ~(size_t)255;
    return p;
  };
  _Float16* tN         = (_Float16*)carve((size_t)BS * DIM * sizeof(_Float16)); // 12.6 MB
  _Float16* cN         = (_Float16*)carve((size_t)BS * DIM * sizeof(_Float16)); // 12.6 MB
  float*    rowPartial = (float*)carve((size_t)NBLK * BS * sizeof(float));      // 2 MB
  float*    colPartial = (float*)carve((size_t)NBLK * BS * sizeof(float));      // 2 MB
  float*    diag       = (float*)carve((size_t)BS * sizeof(float));
  float*    lossArr    = (float*)carve((size_t)BS * sizeof(float));

  nrm_kernel<<<BS, 256, 0, stream>>>(text, tN);
  nrm_kernel<<<BS, 256, 0, stream>>>(ctr, cN);
  gemm_kernel<<<dim3(NBLK, NBLK), 256, 0, stream>>>(tN, cN, rowPartial, colPartial, diag);
  loss_kernel<<<BS / 256, 256, 0, stream>>>(rowPartial, colPartial, diag, lossArr);
  reduce_kernel<<<1, 256, 0, stream>>>(lossArr, out);
}
